// DCTC_6854767804743
// MI455X (gfx1250) — matrix operand order repaired and checked by reading, not compiled, not run on hardware
//
#include <hip/hip_runtime.h>
#include <hip/hip_bf16.h>

typedef __attribute__((ext_vector_type(16))) _Float16 v16h;
typedef __attribute__((ext_vector_type(8)))  _Float16 v8h;
typedef __attribute__((ext_vector_type(8)))  float    v8f;

#define NEGV  (-1e30f)
#define NEGT  (-0.5e30f)
#define TMAX  128
#define SPAD  52
#define KPAD  64
#define CPAD  32

__device__ __forceinline__ float lae(float a, float b) {
    float mx = fmaxf(a, b);
    if (mx <= NEGT) return NEGV;
    float mn = fminf(a, b);
    return mx + log1pf(__expf(mn - mx));
}

// Kernel 1: lse[row] = logsumexp_v(2*logits[row, v]), one streaming pass (online).
__global__ void __launch_bounds__(256) lse_kernel(const float* __restrict__ logits,
                                                  float* __restrict__ lse, int V) {
    size_t row = blockIdx.x;
    const float* p = logits + row * (size_t)V;
    float m = -3.0e38f, s = 0.f;
    for (int i = threadIdx.x; i < V; i += 256) {
        float x = 2.0f * p[i];
        float nm = fmaxf(m, x);
        s = s * __expf(m - nm) + __expf(x - nm);
        m = nm;
    }
    __shared__ float sm_[256], ss_[256];
    sm_[threadIdx.x] = m; ss_[threadIdx.x] = s;
    __syncthreads();
    for (int off = 128; off > 0; off >>= 1) {
        if (threadIdx.x < off) {
            float m1 = sm_[threadIdx.x], s1 = ss_[threadIdx.x];
            float m2 = sm_[threadIdx.x + off], s2 = ss_[threadIdx.x + off];
            float nm = fmaxf(m1, m2);
            sm_[threadIdx.x] = nm;
            ss_[threadIdx.x] = s1 * __expf(m1 - nm) + s2 * __expf(m2 - nm);
        }
        __syncthreads();
    }
    if (threadIdx.x == 0) lse[row] = sm_[0] + __logf(ss_[0]);
}

// Kernel 2: per-batch CTC fwd/bwd + WMMA grouped class sums + argmin + margin CE.
__global__ void __launch_bounds__(256) ctc_kernel(const float* __restrict__ logits,
                                                  const int* __restrict__ targets,
                                                  const int* __restrict__ tlen,
                                                  const float* __restrict__ vrat,
                                                  const float* __restrict__ lse,
                                                  float* __restrict__ nll_out,
                                                  int T, int B, int V, int L) {
    const int b = blockIdx.x;
    const int tid = threadIdx.x;
    const int S = 2 * L + 1;

    __shared__ float     lp[TMAX][SPAD];
    __shared__ float     al[TMAX][SPAD];
    __shared__ __align__(16) _Float16 Em[TMAX][KPAD];   // E matrix, row-contiguous
    __shared__ __align__(16) _Float16 Gt[CPAD][KPAD];   // G transposed: Gt[c][k]
    __shared__ float     Rm[TMAX][CPAD];
    __shared__ float     barv[SPAD], phiv[SPAD];
    __shared__ int       ext_sh[SPAD];
    __shared__ int       skip_sh[SPAD];
    __shared__ int       col_of[SPAD];
    __shared__ int       candv[CPAD];
    __shared__ int       cands[CPAD];
    __shared__ int       ncand_sh, il_sh, idx_sh;
    __shared__ float     end_sh, ctc_sh;
    __shared__ float     red[256];

    if (tid == 0) {
        for (int s = 0; s < S; ++s)
            ext_sh[s] = (s & 1) ? targets[b * L + (s >> 1)] : 0;
        for (int s = 0; s < S; ++s) {
            int e = ext_sh[s];
            skip_sh[s] = (e != 0) && (s < 2 || e != ext_sh[s - 2]);
        }
        int nc = 0;
        candv[nc++] = 0;  // blank
        for (int l = 0; l < L; ++l) {
            int v = targets[b * L + l];
            bool seen = false;
            for (int j = 0; j < nc; ++j) if (candv[j] == v) { seen = true; break; }
            if (!seen && nc < CPAD) candv[nc++] = v;
        }
        for (int i = 1; i < nc; ++i) {          // sort ascending (argmin tie-break by v)
            int key = candv[i], j = i - 1;
            while (j >= 0 && candv[j] > key) { candv[j + 1] = candv[j]; --j; }
            candv[j + 1] = key;
        }
        for (int j = nc; j < CPAD; ++j) candv[j] = -1;
        for (int s = 0; s < S; ++s) {
            int e = ext_sh[s], c = 0;
            for (; c < nc; ++c) if (candv[c] == e) break;
            col_of[s] = c;
        }
        for (int c = 0; c < nc; ++c) {
            int rs = 0;
            for (int s = 0; s < S; ++s) if (ext_sh[s] == candv[c]) { rs = s; break; }
            cands[c] = rs;
        }
        for (int c = nc; c < CPAD; ++c) cands[c] = 0;
        ncand_sh = nc;
        int il = (int)ceilf((float)T * vrat[b]);
        il_sh = il < 1 ? 1 : (il > T ? T : il);
        idx_sh = 2 * tlen[b];
    }
    __syncthreads();
    const int IL = il_sh;

    for (int i = tid; i < TMAX * KPAD; i += 256) Em[i / KPAD][i % KPAD] = (_Float16)0.f;
    for (int i = tid; i < CPAD * KPAD; i += 256) {
        int c = i / KPAD, s = i % KPAD;
        Gt[c][s] = (_Float16)((s < S && col_of[s] == c) ? 1.f : 0.f);
    }
    for (int i = tid; i < IL * S; i += 256) {
        int t = i / S, s = i % S;
        size_t off = ((size_t)t * B + b) * (size_t)V + ext_sh[s];
        lp[t][s] = 2.0f * logits[off] - lse[(size_t)t * B + b];
    }
    __syncthreads();

    // forward alpha
    if (tid < S) al[0][tid] = (tid < 2) ? lp[0][tid] : NEGV;
    __syncthreads();
    for (int t = 1; t < IL; ++t) {
        if (tid < S) {
            int s = tid;
            float a = al[t - 1][s];
            if (s >= 1) a = lae(a, al[t - 1][s - 1]);
            if (s >= 2 && skip_sh[s]) a = lae(a, al[t - 1][s - 2]);
            al[t][s] = a + lp[t][s];
        }
        __syncthreads();
    }
    if (tid == 0) {
        float e = lae(al[IL - 1][idx_sh], al[IL - 1][idx_sh - 1]);
        end_sh = e;
        ctc_sh = (e < NEGT) ? 0.f : -e;   // zero_infinity
    }
    __syncthreads();

    // backward bar: E[t][s] = exp(d(end)/d alpha_t[s] in log domain)
    const bool dead = (end_sh < NEGT);
    if (tid < S) {
        int s = tid;
        float v = (!dead && (s == idx_sh || s == idx_sh - 1)) ? (al[IL - 1][s] - end_sh) : NEGV;
        barv[s] = v;
        Em[IL - 1][s] = (_Float16)((v < NEGT) ? 0.f : __expf(v));
    }
    __syncthreads();
    for (int t = IL - 1; t >= 1; --t) {
        if (tid < S) {
            int s = tid;
            float bb = barv[s], aa = al[t][s];
            phiv[s] = (bb < NEGT || aa < NEGT) ? NEGV : (bb + lp[t][s] - aa);
        }
        __syncthreads();
        if (tid < S) {
            int s = tid;
            float acc = phiv[s];
            if (s + 1 < S) acc = lae(acc, phiv[s + 1]);
            if (s + 2 < S && skip_sh[s + 2]) acc = lae(acc, phiv[s + 2]);
            float ap = al[t - 1][s];
            float nb = (ap < NEGT || acc < NEGT) ? NEGV : (ap + acc);
            barv[s] = nb;
            Em[t - 1][s] = (_Float16)((nb < NEGT) ? 0.f : __expf(nb));
        }
        __syncthreads();
    }

    // WMMA grouped class sums: R[128x32] = E[128x64] @ G[64x32], f16 in / f32 out.
    // A-tile lane layout (ISA 7.12.2, 16-bit A 16x32): lane m<16 holds K 0-7,16-23;
    // lanes 16-31 hold K 8-15,24-31 -> two contiguous 16B chunks per Em row.
    // B-tile: lane holds column n0+m, same K order as A -> Gt row chunks.
    {
        const int wave = tid >> 5;
        const int lane = tid & 31;
        const int mcol = lane & 15;
        const bool lo = (lane < 16);
        const int abase = lo ? 0 : 8;    // half-index offset within a 16-half group
        const int bbase = lo ? 0 : 8;
        for (int task = wave; task < 16; task += 8) {   // uniform per wave, EXEC all ones
            const int t0 = (task & 7) * 16;
            const int n0 = (task >> 3) * 16;
            v8f c = {};
#pragma unroll
            for (int kb = 0; kb < 64; kb += 32) {
                v8h alo = *(const v8h*)&Em[t0 + mcol][kb + abase];
                v8h ahi = *(const v8h*)&Em[t0 + mcol][kb + 16 + abase];
                v16h a = __builtin_shufflevector(alo, ahi,
                        0, 1, 2, 3, 4, 5, 6, 7, 8, 9, 10, 11, 12, 13, 14, 15);
                v8h blo = *(const v8h*)&Gt[n0 + mcol][kb + bbase];
                v8h bhi = *(const v8h*)&Gt[n0 + mcol][kb + bbase + 16];
                v16h bm = __builtin_shufflevector(blo, bhi,
                        0, 1, 2, 3, 4, 5, 6, 7, 8, 9, 10, 11, 12, 13, 14, 15);
                c = __builtin_amdgcn_wmma_f32_16x16x32_f16(false, a, false, bm,
                                                           (short)0, c, false, false);
            }
#pragma unroll
            for (int r = 0; r < 8; ++r)
                Rm[t0 + r + (lo ? 0 : 8)][n0 + mcol] = c[r];
        }
    }
    __syncthreads();

    // argmin over candidates (grad/prob), then margin CE
    float mysel = 0.f;
    if (tid < IL) {
        const int t = tid;
        const int nc = ncand_sh;
        float bestr = 3.4e38f;
        int bestc = 0;
        for (int c = 0; c < nc; ++c) {
            float prob = __expf(lp[t][cands[c]]);
            float ratio = -Rm[t][c] / fmaxf(prob, 1e-8f);   // grad = -R, R >= 0
            if (ratio < bestr) { bestr = ratio; bestc = c; }
        }
        if (bestr >= 0.f) bestc = 0;                        // all-zero row -> argmin = v0 = blank
        float lpc = lp[t][cands[bestc]];
        float prob = __expf(lpc);
        const float sm = 0.70f;                             // S_SCALE * MARGIN
        const float exm1 = __expf(-sm) - 1.f;
        mysel = sm + log1pf(prob * exm1) - lpc;             // == -(logit_c - sm - lse_adj)
    }
    red[tid] = mysel;
    __syncthreads();
    for (int off = 128; off > 0; off >>= 1) {
        if (tid < off) red[tid] += red[tid + off];
        __syncthreads();
    }
    if (tid == 0) nll_out[b] = 0.1f * red[0] + 1.0f * ctc_sh;   // ALPHA*ce + BETA*ctc
}

// Kernel 3: deterministic mean over B per-sample losses.
__global__ void __launch_bounds__(256) finalize_kernel(const float* __restrict__ nll,
                                                       float* __restrict__ out, int B) {
    __shared__ float r[256];
    float acc = 0.f;
    for (int i = threadIdx.x; i < B; i += 256) acc += nll[i];
    r[threadIdx.x] = acc;
    __syncthreads();
    for (int off = 128; off > 0; off >>= 1) {
        if (threadIdx.x < off) r[threadIdx.x] += r[threadIdx.x + off];
        __syncthreads();
    }
    if (threadIdx.x == 0) out[0] = r[0] / (float)B;
}

extern "C" void kernel_launch(void* const* d_in, const int* in_sizes, int n_in,
                              void* d_out, int out_size, void* d_ws, size_t ws_size,
                              hipStream_t stream) {
    const float* logits  = (const float*)d_in[0];
    const int*   targets = (const int*)d_in[1];
    const int*   tlen    = (const int*)d_in[2];
    const float* vrat    = (const float*)d_in[3];

    const int B = in_sizes[2];
    const int L = in_sizes[1] / B;
    const int T = 128;                               // reference shape
    const int V = (int)(((long long)in_sizes[0] / B) / T);

    float* lse = (float*)d_ws;                       // T*B floats
    float* nll = lse + (size_t)T * B;                // B floats

    lse_kernel<<<T * B, 256, 0, stream>>>(logits, lse, V);
    ctc_kernel<<<B, 256, 0, stream>>>(logits, targets, tlen, vrat, lse, nll, T, B, V, L);
    finalize_kernel<<<1, 256, 0, stream>>>(nll, (float*)d_out, B);
}
